// CRGainHopLayerAnnealed_89163521065431
// MI455X (gfx1250) — matrix operand order repaired and checked by reading, not compiled, not run on hardware
//
#include <hip/hip_runtime.h>
#include <hip/hip_bf16.h>
#include <math.h>

typedef __attribute__((ext_vector_type(16))) __bf16 v16bf;
typedef __attribute__((ext_vector_type(8)))  __bf16 v8bf;
typedef __attribute__((ext_vector_type(4)))  __bf16 v4bf;
typedef __attribute__((ext_vector_type(8)))  float  v8f;
typedef __attribute__((ext_vector_type(4)))  float  v4f;

#define NROWS 8192
#define FIN   256
#define FOUT  128
#define DSUB  64
#define KHOP  3
#define MTILE 32
#define ASTRIDE 40   // padded LDS row stride (bf16 elems): 20 words -> conflict-free frags

#define COEFF_CR (64.0f / (8192.0f * 0.5f * 0.5f))   // d/(n*eps^2) = 0.03125
#define ETA_C    0.5f
#define LAM_C    0.3f

// ---- workspace layout (float offsets) ----
constexpr size_t OFF_Z  = 0;                                        // K*N*D
constexpr size_t OFF_B  = OFF_Z + (size_t)KHOP * NROWS * DSUB;      // N*FOUT
constexpr size_t OFF_E  = OFF_B + (size_t)NROWS * FOUT;             // N*FOUT
constexpr size_t OFF_G  = OFF_E + (size_t)NROWS * FOUT;             // K*D*D
constexpr size_t OFF_I  = OFF_G + (size_t)KHOP * DSUB * DSUB;       // K*D*D (Minv)
constexpr size_t OFF_Q  = OFF_I + (size_t)KHOP * DSUB * DSUB;       // K*D*FOUT
constexpr size_t OFF_W  = OFF_Q + (size_t)KHOP * DSUB * FOUT;       // w[3]+pad
constexpr size_t OFF_BT = OFF_W + 8;   // bf16 region: Bt_hi[FOUT*N], Bt_lo[FOUT*N]

constexpr size_t OUT_TAIL = (size_t)NROWS * FOUT;  // 1048576

// ---------------- Z[k] = hop[k] @ W_stack[k]^T  -> [K, N, D] ----------------
__global__ void z_kernel(const float* __restrict__ hop,
                         const float* __restrict__ Wstack,
                         float* __restrict__ ws) {
  __shared__ float sh[32 * FIN];                       // 32 KB
  const int k  = blockIdx.y;
  const int n0 = blockIdx.x * 32;
  const float* hk = hop + (size_t)k * NROWS * FIN;
  for (int i = threadIdx.x; i < 32 * FIN; i += blockDim.x)
    sh[i] = hk[(size_t)(n0 + (i >> 8)) * FIN + (i & 255)];
  __syncthreads();
  const float* Wk = Wstack + (size_t)k * DSUB * FIN;
  float* Z = ws + OFF_Z + (size_t)k * NROWS * DSUB;
  for (int o = threadIdx.x; o < 32 * DSUB; o += blockDim.x) {
    const int r = o >> 6, dd = o & 63;
    const float* wrow = Wk + (size_t)dd * FIN;
    const float* hrow = sh + r * FIN;
    float acc = 0.f;
    for (int f = 0; f < FIN; f += 4) {
      v4f wv = *(const v4f*)(wrow + f);
      acc += hrow[f] * wv[0] + hrow[f + 1] * wv[1] +
             hrow[f + 2] * wv[2] + hrow[f + 3] * wv[3];
    }
    Z[(size_t)(n0 + r) * DSUB + dd] = acc;
  }
}

// ---- B = H @ W_out^T; also emit transposed bf16 hi/lo splits Bt[FOUT][N] ----
__global__ void b_kernel(const float* __restrict__ H,
                         const float* __restrict__ Wout,
                         float* __restrict__ ws) {
  __shared__ float sh[32 * FIN];
  const int n0 = blockIdx.x * 32;
  for (int i = threadIdx.x; i < 32 * FIN; i += blockDim.x)
    sh[i] = H[(size_t)(n0 + (i >> 8)) * FIN + (i & 255)];
  __syncthreads();
  float* B = ws + OFF_B;
  __bf16* bth = (__bf16*)(ws + OFF_BT);
  __bf16* btl = bth + (size_t)FOUT * NROWS;
  for (int o = threadIdx.x; o < 32 * FOUT; o += blockDim.x) {
    const int r = o >> 7, c = o & 127;
    const float* wrow = Wout + (size_t)c * FIN;
    const float* hrow = sh + r * FIN;
    float acc = 0.f;
    for (int f = 0; f < FIN; f += 4) {
      v4f wv = *(const v4f*)(wrow + f);
      acc += hrow[f] * wv[0] + hrow[f + 1] * wv[1] +
             hrow[f + 2] * wv[2] + hrow[f + 3] * wv[3];
    }
    const int n = n0 + r;
    B[(size_t)n * FOUT + c] = acc;
    __bf16 hi = (__bf16)acc;
    bth[(size_t)c * NROWS + n] = hi;
    btl[(size_t)c * NROWS + n] = (__bf16)(acc - (float)hi);
  }
}

// ---------------- zero the Gram accumulators ----------------
__global__ void zero_g_kernel(float* __restrict__ ws) {
  int i = blockIdx.x * blockDim.x + threadIdx.x;
  if (i < KHOP * DSUB * DSUB) ws[OFF_G + i] = 0.f;
}

// ---------------- G[k] += Zchunk^T @ Zchunk (atomic reduce) ----------------
__global__ void gram_kernel(float* __restrict__ ws) {
  __shared__ float zs[128 * DSUB];                     // 32 KB
  const int k = blockIdx.y, n0 = blockIdx.x * 128;
  const float* Z = ws + OFF_Z + (size_t)k * NROWS * DSUB;
  for (int i = threadIdx.x; i < 128 * DSUB; i += blockDim.x)
    zs[i] = Z[(size_t)n0 * DSUB + i];
  __syncthreads();
  float* G = ws + OFF_G + (size_t)k * DSUB * DSUB;
  for (int o = threadIdx.x; o < DSUB * DSUB; o += blockDim.x) {
    const int a = o >> 6, b = o & 63;
    float acc = 0.f;
    for (int r = 0; r < 128; ++r) acc += zs[r * DSUB + a] * zs[r * DSUB + b];
    atomicAdd(&G[o], acc);
  }
}

// ---- one block: M=I+coeff*G, Cholesky, R, Minv, delta_R, softmax weights ----
__global__ void stats_kernel(const float* __restrict__ tau_p,
                             float* __restrict__ ws,
                             float* __restrict__ out) {
  __shared__ float a[DSUB][DSUB + 1];
  __shared__ float x[DSUB][DSUB + 1];
  __shared__ float Rs[KHOP];
  const int tid = threadIdx.x;
  for (int k = 0; k < KHOP; ++k) {
    const float* G = ws + OFF_G + (size_t)k * DSUB * DSUB;
    for (int i = tid; i < DSUB * DSUB; i += blockDim.x) {
      const int r = i >> 6, c = i & 63;
      a[r][c] = (r == c ? 1.f : 0.f) + COEFF_CR * G[i];
    }
    __syncthreads();
    // in-place lower Cholesky
    for (int j = 0; j < DSUB; ++j) {
      if (tid == 0) a[j][j] = sqrtf(a[j][j]);
      __syncthreads();
      for (int r = j + 1 + tid; r < DSUB; r += blockDim.x) a[r][j] /= a[j][j];
      __syncthreads();
      const int m = DSUB - 1 - j;
      for (int p = tid; p < m * m; p += blockDim.x) {
        const int r = j + 1 + p / m, c = j + 1 + p % m;
        if (c <= r) a[r][c] -= a[r][j] * a[c][j];
      }
      __syncthreads();
    }
    if (tid == 0) {
      float s = 0.f;
      for (int j = 0; j < DSUB; ++j) s += logf(a[j][j]);
      Rs[k] = s;                                       // 0.5*logdet(M)
    }
    // Minv columns via fwd/back substitution (one column per thread)
    if (tid < DSUB) {
      const int c = tid;
      for (int i = 0; i < DSUB; ++i) {
        float s = (i == c) ? 1.f : 0.f;
        for (int j = 0; j < i; ++j) s -= a[i][j] * x[j][c];
        x[i][c] = s / a[i][i];
      }
      for (int i = DSUB - 1; i >= 0; --i) {
        float s = x[i][c];
        for (int j = i + 1; j < DSUB; ++j) s -= a[j][i] * x[j][c];
        x[i][c] = s / a[i][i];
      }
    }
    __syncthreads();
    float* Minv = ws + OFF_I + (size_t)k * DSUB * DSUB;
    for (int i = tid; i < DSUB * DSUB; i += blockDim.x) Minv[i] = x[i >> 6][i & 63];
    __syncthreads();
  }
  if (tid == 0) {
    const float d0 = Rs[0], d1 = Rs[1] - Rs[0], d2 = Rs[2] - Rs[1];
    const float mean = (d0 + d1 + d2) * (1.f / 3.f);
    const float var = ((d0 - mean) * (d0 - mean) + (d1 - mean) * (d1 - mean) +
                       (d2 - mean) * (d2 - mean)) * 0.5f;   // ddof=1
    const float sd = sqrtf(var) + 1e-6f;
    const float n0 = (d0 - mean) / sd, n1 = (d1 - mean) / sd, n2 = (d2 - mean) / sd;
    const float inv_tau = 1.f / tau_p[0];
    const float e0 = n0 * inv_tau, e1 = n1 * inv_tau, e2 = n2 * inv_tau;
    const float mx = fmaxf(e0, fmaxf(e1, e2));
    const float x0 = expf(e0 - mx), x1 = expf(e1 - mx), x2 = expf(e2 - mx);
    const float inv = 1.f / (x0 + x1 + x2);
    const float w0 = x0 * inv, w1 = x1 * inv, w2 = x2 * inv;
    out[OUT_TAIL + 0] = w0; out[OUT_TAIL + 1] = w1; out[OUT_TAIL + 2] = w2;
    out[OUT_TAIL + 3] = n0; out[OUT_TAIL + 4] = n1; out[OUT_TAIL + 5] = n2;
    out[OUT_TAIL + 6] = d0; out[OUT_TAIL + 7] = d1; out[OUT_TAIL + 8] = d2;
    ws[OFF_W + 0] = w0; ws[OFF_W + 1] = w1; ws[OFF_W + 2] = w2;
  }
}

// ---- Q[k] = (eta*coeff*w_k) * Minv_k @ (W_k @ W_out^T)  -> [K, D, FOUT] ----
__global__ void q_kernel(const float* __restrict__ Wstack,
                         const float* __restrict__ Wout,
                         float* __restrict__ ws) {
  __shared__ float Ts[DSUB * FOUT];                    // 32 KB
  __shared__ float Mi[DSUB * DSUB];                    // 16 KB
  const int k = blockIdx.x;
  const float* Wk = Wstack + (size_t)k * DSUB * FIN;
  for (int o = threadIdx.x; o < DSUB * FOUT; o += blockDim.x) {
    const int dd = o >> 7, c = o & 127;
    float acc = 0.f;
    for (int f = 0; f < FIN; f += 4) {
      v4f wa = *(const v4f*)(Wk + (size_t)dd * FIN + f);
      v4f wb = *(const v4f*)(Wout + (size_t)c * FIN + f);
      acc += wa[0] * wb[0] + wa[1] * wb[1] + wa[2] * wb[2] + wa[3] * wb[3];
    }
    Ts[o] = acc;
  }
  for (int i = threadIdx.x; i < DSUB * DSUB; i += blockDim.x)
    Mi[i] = ws[OFF_I + (size_t)k * DSUB * DSUB + i];
  __syncthreads();
  const float scale = ETA_C * COEFF_CR * ws[OFF_W + k];
  float* Q = ws + OFF_Q + (size_t)k * DSUB * FOUT;
  for (int o = threadIdx.x; o < DSUB * FOUT; o += blockDim.x) {
    const int dd = o >> 7, c = o & 127;
    float acc = 0.f;
    for (int j = 0; j < DSUB; ++j) acc += Mi[dd * DSUB + j] * Ts[j * FOUT + c];
    Q[o] = scale * acc;
  }
}

// ---------------- E = B + sum_k Z_k @ Q_k  -> [N, FOUT] ----------------
__global__ void e_kernel(float* __restrict__ ws) {
  __shared__ float zr[32 * KHOP * DSUB];               // 24 KB
  const int n0 = blockIdx.x * 32;
  for (int i = threadIdx.x; i < 32 * KHOP * DSUB; i += blockDim.x) {
    const int r = i / (KHOP * DSUB), kd = i % (KHOP * DSUB);
    const int k = kd >> 6, dd = kd & 63;
    zr[i] = ws[OFF_Z + ((size_t)k * NROWS + n0 + r) * DSUB + dd];
  }
  __syncthreads();
  const float* Q = ws + OFF_Q;
  for (int o = threadIdx.x; o < 32 * FOUT; o += blockDim.x) {
    const int r = o >> 7, c = o & 127, n = n0 + r;
    float acc = ws[OFF_B + (size_t)n * FOUT + c];
    for (int kd = 0; kd < KHOP * DSUB; ++kd)
      acc += zr[r * (KHOP * DSUB) + kd] * Q[(size_t)kd * FOUT + c];
    ws[OFF_E + (size_t)n * FOUT + c] = acc;
  }
}

// ---- main kernel: H_pre = E - 0.15*(L@B) via split-bf16 WMMA ----
// Block: 32 rows x 128 cols. L tile staged once per block to LDS (hi/lo bf16),
// waves own distinct 16x16 C tiles (2 each), interleaved WMMA accumulators.
__global__ void __launch_bounds__(256)
big_kernel(const float* __restrict__ Lm, const float* __restrict__ thr,
           const float* __restrict__ gamma, const float* __restrict__ beta,
           const float* __restrict__ ws, float* __restrict__ out) {
  __shared__ __bf16 sahi[MTILE * ASTRIDE];             // 2.5 KB
  __shared__ __bf16 salo[MTILE * ASTRIDE];             // 2.5 KB
  __shared__ float sm[MTILE * FOUT];                   // 16 KB
  __shared__ float mu_s[MTILE], rs_s[MTILE];

  const int tid = threadIdx.x;
  const int m0 = blockIdx.x * MTILE;
  const int wave = tid >> 5, lane = tid & 31;
  const int msub = wave >> 2;                 // 0/1: which 16-row half
  const int col0 = (wave & 3) * 32 + (lane & 15);
  const int col1 = col0 + 16;
  const int kb   = (lane & 16) ? 16 : 0;      // A-frag K sub-block (contiguous, matches B)
  const int koff = (lane & 16) ? 16 : 0;      // B-frag K sub-block
  const int arow_l = msub * 16 + (lane & 15); // A-frag row within tile

  // tile loader assignment: thread -> (row, 4-col group)
  const int ldr = tid >> 3;
  const int ldc = (tid & 7) * 4;
  const float* gload = Lm + (size_t)(m0 + ldr) * NROWS + ldc;

  const __bf16* bth = (const __bf16*)(ws + OFF_BT);
  const __bf16* btl = bth + (size_t)FOUT * NROWS;
  const __bf16* b0h = bth + (size_t)col0 * NROWS + koff;
  const __bf16* b0l = btl + (size_t)col0 * NROWS + koff;
  const __bf16* b1h = bth + (size_t)col1 * NROWS + koff;
  const __bf16* b1l = btl + (size_t)col1 * NROWS + koff;

  v8f c0, c1;
  #pragma unroll
  for (int i = 0; i < 8; ++i) { c0[i] = 0.f; c1[i] = 0.f; }

  // prologue: load+convert+stage tile 0
  {
    v4f fr = *(const v4f*)(gload);
    v4bf h4, l4;
    #pragma unroll
    for (int e = 0; e < 4; ++e) {
      __bf16 h = (__bf16)fr[e];
      h4[e] = h; l4[e] = (__bf16)(fr[e] - (float)h);
    }
    *(v4bf*)&sahi[ldr * ASTRIDE + ldc] = h4;
    *(v4bf*)&salo[ldr * ASTRIDE + ldc] = l4;
  }
  __syncthreads();

  for (int k0 = 0; k0 < NROWS; k0 += 32) {
    // prefetch next L tile into registers (no LDS dependence)
    v4f nxt;
    const bool more = (k0 + 32) < NROWS;
    if (more) nxt = *(const v4f*)(gload + k0 + 32);

    // A fragments from LDS (two contiguous 8-elem runs per lane)
    v8bf ah0 = *(const v8bf*)&sahi[arow_l * ASTRIDE + kb];
    v8bf ah1 = *(const v8bf*)&sahi[arow_l * ASTRIDE + kb + 8];
    v8bf al0 = *(const v8bf*)&salo[arow_l * ASTRIDE + kb];
    v8bf al1 = *(const v8bf*)&salo[arow_l * ASTRIDE + kb + 8];
    v16bf ah = __builtin_shufflevector(ah0, ah1, 0,1,2,3,4,5,6,7,8,9,10,11,12,13,14,15);
    v16bf al = __builtin_shufflevector(al0, al1, 0,1,2,3,4,5,6,7,8,9,10,11,12,13,14,15);

    // B fragments: pre-transposed Bt -> 16 contiguous K values per lane
    v16bf b0hh = *(const v16bf*)(b0h + k0);
    v16bf b0ll = *(const v16bf*)(b0l + k0);
    v16bf b1hh = *(const v16bf*)(b1h + k0);
    v16bf b1ll = *(const v16bf*)(b1l + k0);

    // split product Lhi*Bhi + Lhi*Blo + Llo*Bhi; interleave the two
    // accumulators so dependent WMMAs are distance-2 (hides D->C latency)
    c0 = __builtin_amdgcn_wmma_f32_16x16x32_bf16(false, ah, false, b0hh, (short)0, c0, false, false);
    c1 = __builtin_amdgcn_wmma_f32_16x16x32_bf16(false, ah, false, b1hh, (short)0, c1, false, false);
    c0 = __builtin_amdgcn_wmma_f32_16x16x32_bf16(false, ah, false, b0ll, (short)0, c0, false, false);
    c1 = __builtin_amdgcn_wmma_f32_16x16x32_bf16(false, ah, false, b1ll, (short)0, c1, false, false);
    c0 = __builtin_amdgcn_wmma_f32_16x16x32_bf16(false, al, false, b0hh, (short)0, c0, false, false);
    c1 = __builtin_amdgcn_wmma_f32_16x16x32_bf16(false, al, false, b1hh, (short)0, c1, false, false);

    if (more) {
      // convert prefetched tile and re-stage LDS
      v4bf h4, l4;
      #pragma unroll
      for (int e = 0; e < 4; ++e) {
        __bf16 h = (__bf16)nxt[e];
        h4[e] = h; l4[e] = (__bf16)(nxt[e] - (float)h);
      }
      __syncthreads();   // all frag reads of current tile done
      *(v4bf*)&sahi[ldr * ASTRIDE + ldc] = h4;
      *(v4bf*)&salo[ldr * ASTRIDE + ldc] = l4;
      __syncthreads();   // stores visible
    }
  }

  // epilogue: x = E - eta*lambda*(L@B); soft-threshold into LDS tile
  const float thr0 = fabsf(thr[col0]);
  const float thr1 = fabsf(thr[col1]);
  const float* E = ws + OFF_E;
  #pragma unroll
  for (int i = 0; i < 8; ++i) {
    const int r = msub * 16 + ((lane < 16) ? i : (i + 8));  // C/D VGPR layout
    const float x0 = E[(size_t)(m0 + r) * FOUT + col0] - (ETA_C * LAM_C) * c0[i];
    const float a0 = fabsf(x0) - thr0;
    sm[r * FOUT + col0] = (a0 > 0.f) ? copysignf(a0, x0) : 0.f;
    const float x1 = E[(size_t)(m0 + r) * FOUT + col1] - (ETA_C * LAM_C) * c1[i];
    const float a1 = fabsf(x1) - thr1;
    sm[r * FOUT + col1] = (a1 > 0.f) ? copysignf(a1, x1) : 0.f;
  }
  __syncthreads();
  if (tid < MTILE) {
    float s = 0.f, s2 = 0.f;
    for (int j = 0; j < FOUT; ++j) {
      const float v = sm[tid * FOUT + j];
      s += v; s2 += v * v;
    }
    const float mu = s * (1.f / FOUT);
    mu_s[tid] = mu;
    rs_s[tid] = rsqrtf(s2 * (1.f / FOUT) - mu * mu + 1e-5f);
  }
  __syncthreads();
  for (int idx = tid; idx < MTILE * FOUT; idx += 256) {
    const int r = idx >> 7, j = idx & 127;
    out[(size_t)(m0 + r) * FOUT + j] =
        (sm[idx] - mu_s[r]) * rs_s[r] * gamma[j] + beta[j];
  }
}

extern "C" void kernel_launch(void* const* d_in, const int* in_sizes, int n_in,
                              void* d_out, int out_size, void* d_ws, size_t ws_size,
                              hipStream_t stream) {
  (void)in_sizes; (void)n_in; (void)out_size; (void)ws_size;
  const float* H      = (const float*)d_in[0];
  const float* hop    = (const float*)d_in[1];
  const float* Lm     = (const float*)d_in[2];
  const float* Wstack = (const float*)d_in[3];
  const float* Wout   = (const float*)d_in[4];
  const float* thr    = (const float*)d_in[5];
  const float* gamma  = (const float*)d_in[6];
  const float* beta   = (const float*)d_in[7];
  const float* tau    = (const float*)d_in[8];
  float* ws  = (float*)d_ws;
  float* out = (float*)d_out;

  z_kernel     <<<dim3(NROWS / 32, KHOP), 256, 0, stream>>>(hop, Wstack, ws);
  b_kernel     <<<dim3(NROWS / 32),       256, 0, stream>>>(H, Wout, ws);
  zero_g_kernel<<<dim3(48),               256, 0, stream>>>(ws);
  gram_kernel  <<<dim3(NROWS / 128, KHOP),256, 0, stream>>>(ws);
  stats_kernel <<<dim3(1),                256, 0, stream>>>(tau, ws, out);
  q_kernel     <<<dim3(KHOP),             256, 0, stream>>>(Wstack, Wout, ws);
  e_kernel     <<<dim3(NROWS / 32),       256, 0, stream>>>(ws);
  big_kernel   <<<dim3(NROWS / MTILE),    256, 0, stream>>>(Lm, thr, gamma, beta, ws, out);
}
